// SceneGraphGAT_84774064489189
// MI455X (gfx1250) — matrix operand order repaired and checked by reading, not compiled, not run on hardware
//
#include <hip/hip_runtime.h>
#include <stdint.h>

// Problem constants (fixed by the reference)
#define NN    50000
#define EE    800000
#define BB    256
#define HH    4
#define DD    32
#define HID   128
#define EMBD  128
#define ETOT  (EE + NN)   // edges + self loops = 850000

typedef __attribute__((ext_vector_type(16))) __bf16 v16bf;
typedef __attribute__((ext_vector_type(8)))  float  v8f;

union AFrag { uint4 u[2]; v16bf v; };

__device__ __forceinline__ unsigned short f32_to_bf16(float f) {
  unsigned u = __float_as_uint(f);
  u += 0x7FFFu + ((u >> 16) & 1u);         // round-to-nearest-even
  return (unsigned short)(u >> 16);
}

// ---------------------------------------------------------------- utilities
__global__ void zero_kernel(float* __restrict__ p, int n) {
  int i = blockIdx.x * blockDim.x + threadIdx.x;
  if (i < n) p[i] = 0.0f;
}

// embed gather -> bf16 activations (N x 128)
__global__ void gather_embed_kernel(const int* __restrict__ labels,
                                    const float* __restrict__ embed,
                                    unsigned short* __restrict__ Xb) {
  int idx = blockIdx.x * blockDim.x + threadIdx.x;
  if (idx >= NN * EMBD) return;
  int n = idx >> 7, c = idx & 127;
  Xb[idx] = f32_to_bf16(embed[(size_t)labels[n] * EMBD + c]);
}

// W (128x128, k-major) -> Wt bf16 (n-major: Wt[n*128+k]) so B-fragment loads are contiguous
__global__ void wt_bf16_kernel(const float* __restrict__ W,
                               unsigned short* __restrict__ Wt) {
  int idx = blockIdx.x * blockDim.x + threadIdx.x;
  if (idx >= HID * HID) return;
  int k = idx >> 7, n = idx & 127;
  Wt[n * HID + k] = f32_to_bf16(W[idx]);
}

// ------------------------------------------------------------ WMMA GEMM
// H = X(N x 128, bf16) @ W(128 x 128, via Wt), f32 accumulate.
// Block = 256 threads = 8 waves; wave w computes the 16x16 tile (rows blockIdx*16.., cols w*16..).
__global__ __launch_bounds__(256) void gemm_bf16_kernel(
    const unsigned short* __restrict__ X,
    const unsigned short* __restrict__ Wt,
    float* __restrict__ Hout) {
  const int lane = threadIdx.x & 31;
  const int wave = threadIdx.x >> 5;        // column tile 0..7
  const int row0 = blockIdx.x * 16;
  const int m    = lane & 15;
  const int hi   = lane >> 4;               // lane group 0/1

  const unsigned short* xrow = X  + (size_t)(row0 + m)      * HID;
  const unsigned short* wrow = Wt + (size_t)(wave * 16 + m) * HID;

  v8f acc = {};
#pragma unroll
  for (int k = 0; k < 128; k += 32) {
    AFrag a, b;
    // A 16x32 bf16 layout: lane(m, hi) elems 0..7 -> K = k+hi*8 .. +7 ; elems 8..15 -> +16
    const int ka = k + hi * 8;
    a.u[0] = *(const uint4*)(xrow + ka);
    a.u[1] = *(const uint4*)(xrow + ka + 16);
    // B 32x16 bf16 layout: lane(n=m, hi) elems 0..7 -> K = k+hi*8 .. +7 ; elems 8..15 -> +16
    const int kb = k + hi * 8;
    b.u[0] = *(const uint4*)(wrow + kb);
    b.u[1] = *(const uint4*)(wrow + kb + 16);
    acc = __builtin_amdgcn_wmma_f32_16x16x32_bf16(false, a.v, false, b.v,
                                                  (short)0, acc, false, false);
  }
  // D layout: col n = lane&15, row = r + 8*hi
  float* out = Hout + (size_t)row0 * HID + wave * 16 + m;
#pragma unroll
  for (int r = 0; r < 8; ++r)
    out[(size_t)(r + 8 * hi) * HID] = acc[r];
}

// -------------------------------------------------- attention coefficients
// alpha_src[n,h] = sum_d h[n,h,d]*a_src[h,d] ; one node per block, one head per wave32
__global__ __launch_bounds__(128) void alpha_kernel(
    const float* __restrict__ Hf,
    const float* __restrict__ a_src, const float* __restrict__ a_dst,
    float* __restrict__ as, float* __restrict__ ad) {
  const int n = blockIdx.x, tid = threadIdx.x;
  const int head = tid >> 5, lane = tid & 31;
  float hv = Hf[(size_t)n * HID + tid];
  float vs = hv * a_src[head * DD + lane];
  float vd = hv * a_dst[head * DD + lane];
#pragma unroll
  for (int off = 16; off; off >>= 1) {
    vs += __shfl_down(vs, off, 32);
    vd += __shfl_down(vd, off, 32);
  }
  if (lane == 0) { as[n * HH + head] = vs; ad[n * HH + head] = vd; }
}

// ------------------------------------------------------------- edge passes
__global__ void edge_logit_max_kernel(const int* __restrict__ ei,
                                      const float* __restrict__ as,
                                      const float* __restrict__ ad,
                                      float* __restrict__ ev,
                                      unsigned int* __restrict__ mbits) {
  int idx = blockIdx.x * blockDim.x + threadIdx.x;
  if (idx >= ETOT * HH) return;
  int e = idx >> 2, h = idx & 3;
  int s = (e < EE) ? ei[e]      : (e - EE);
  int d = (e < EE) ? ei[EE + e] : (e - EE);
  float v = as[s * HH + h] + ad[d * HH + h];
  v = (v > 0.0f) ? v : v * 0.2f;             // leaky relu
  ev[idx] = v;
  unsigned int b = __float_as_uint(v);       // order-preserving uint encoding
  b = (b & 0x80000000u) ? ~b : (b | 0x80000000u);
  atomicMax(&mbits[d * HH + h], b);
}

__global__ void edge_exp_sum_kernel(const int* __restrict__ ei,
                                    const unsigned int* __restrict__ mbits,
                                    float* __restrict__ ev,
                                    float* __restrict__ denom) {
  int idx = blockIdx.x * blockDim.x + threadIdx.x;
  if (idx >= ETOT * HH) return;
  int e = idx >> 2, h = idx & 3;
  int d = (e < EE) ? ei[EE + e] : (e - EE);
  unsigned int mb = mbits[d * HH + h];
  float m = (mb & 0x80000000u) ? __uint_as_float(mb & 0x7FFFFFFFu)
                               : __uint_as_float(~mb);
  float ex = __expf(ev[idx] - m);
  ev[idx] = ex;
  atomicAdd(&denom[d * HH + h], ex);
}

__global__ void edge_aggregate_kernel(const int* __restrict__ ei,
                                      const float* __restrict__ Hf,
                                      const float* __restrict__ ev,
                                      const float* __restrict__ denom,
                                      float* __restrict__ agg) {
  int idx = blockIdx.x * blockDim.x + threadIdx.x;
  if (idx >= ETOT * HID) return;
  int e = idx >> 7, c = idx & 127;
  int h = c >> 5;
  int s = (e < EE) ? ei[e]      : (e - EE);
  int d = (e < EE) ? ei[EE + e] : (e - EE);
  float coef = ev[e * HH + h] / denom[d * HH + h];
  atomicAdd(&agg[(size_t)d * HID + c], Hf[(size_t)s * HID + c] * coef);
}

// ---------------------------------------------- bias + LayerNorm + ReLU
__global__ __launch_bounds__(128) void ln_relu_kernel(
    const float* __restrict__ agg, const float* __restrict__ bias,
    const float* __restrict__ gamma, const float* __restrict__ beta,
    unsigned short* __restrict__ Xb, float* __restrict__ Xf) {
  __shared__ float red[4];
  const int n = blockIdx.x, tid = threadIdx.x;
  const int w = tid >> 5, lane = tid & 31;
  float v = agg[(size_t)n * HID + tid] + bias[tid];
  float s = v;
#pragma unroll
  for (int off = 16; off; off >>= 1) s += __shfl_down(s, off, 32);
  if (lane == 0) red[w] = s;
  __syncthreads();
  const float mu = (red[0] + red[1] + red[2] + red[3]) * (1.0f / 128.0f);
  __syncthreads();
  const float dv = v - mu;
  float q = dv * dv;
#pragma unroll
  for (int off = 16; off; off >>= 1) q += __shfl_down(q, off, 32);
  if (lane == 0) red[w] = q;
  __syncthreads();
  const float var = (red[0] + red[1] + red[2] + red[3]) * (1.0f / 128.0f);
  float y = dv * rsqrtf(var + 1e-5f) * gamma[tid] + beta[tid];
  y = fmaxf(y, 0.0f);
  Xb[(size_t)n * HID + tid] = f32_to_bf16(y);
  Xf[(size_t)n * HID + tid] = y;
}

// ------------------------------------------------------------- mean pool
__global__ void pool_scatter_kernel(const float* __restrict__ X,
                                    const int* __restrict__ batch,
                                    float* __restrict__ sums,
                                    float* __restrict__ cnt) {
  int idx = blockIdx.x * blockDim.x + threadIdx.x;
  if (idx >= NN * HID) return;
  int n = idx >> 7, c = idx & 127;
  int b = batch[n];
  atomicAdd(&sums[(size_t)b * HID + c], X[idx]);
  if (c == 0) atomicAdd(&cnt[b], 1.0f);
}

__global__ void pool_final_kernel(const float* __restrict__ sums,
                                  const float* __restrict__ cnt,
                                  float* __restrict__ out) {
  int idx = blockIdx.x * blockDim.x + threadIdx.x;
  if (idx >= BB * HID) return;
  int b = idx >> 7;
  out[idx] = sums[idx] / fmaxf(cnt[b], 1.0f);
}

// ----------------------------------------------------------------- launch
static inline size_t align256(size_t x) { return (x + 255) & ~(size_t)255; }
static inline int cdiv(int a, int b) { return (a + b - 1) / b; }

extern "C" void kernel_launch(void* const* d_in, const int* in_sizes, int n_in,
                              void* d_out, int out_size, void* d_ws, size_t ws_size,
                              hipStream_t stream) {
  const int*   labels = (const int*)  d_in[0];
  const int*   eidx   = (const int*)  d_in[1];
  const int*   batch  = (const int*)  d_in[2];
  const float* embed  = (const float*)d_in[3];
  const float* W0   = (const float*)d_in[4];
  const float* as0  = (const float*)d_in[5];
  const float* ad0  = (const float*)d_in[6];
  const float* b0   = (const float*)d_in[7];
  const float* g0   = (const float*)d_in[8];
  const float* be0  = (const float*)d_in[9];
  const float* W1   = (const float*)d_in[10];
  const float* as1  = (const float*)d_in[11];
  const float* ad1  = (const float*)d_in[12];
  const float* b1   = (const float*)d_in[13];
  const float* g1   = (const float*)d_in[14];
  const float* be1  = (const float*)d_in[15];

  char* base = (char*)d_ws;
  size_t off = 0;
  unsigned short* XB  = (unsigned short*)(base + off); off += align256((size_t)NN * HID * 2);
  float*          Hf  = (float*)(base + off);          off += align256((size_t)NN * HID * 4);
  float*          AGG = (float*)(base + off);          off += align256((size_t)NN * HID * 4);
  float*          AS  = (float*)(base + off);          off += align256((size_t)NN * HH * 4);
  float*          AD  = (float*)(base + off);          off += align256((size_t)NN * HH * 4);
  unsigned int*   MB  = (unsigned int*)(base + off);   off += align256((size_t)NN * HH * 4);
  float*          DEN = (float*)(base + off);          off += align256((size_t)NN * HH * 4);
  float*          EV  = (float*)(base + off);          off += align256((size_t)ETOT * HH * 4);
  unsigned short* WT  = (unsigned short*)(base + off); off += align256((size_t)HID * HID * 2);
  float*          SUM = (float*)(base + off);          off += align256((size_t)BB * HID * 4);
  float*          CNT = (float*)(base + off);          off += align256((size_t)BB * 4);
  (void)ws_size; (void)n_in; (void)in_sizes; (void)out_size;

  const int T = 256;

  // x = embed[labels] -> bf16
  gather_embed_kernel<<<cdiv(NN * EMBD, T), T, 0, stream>>>(labels, embed, XB);

  for (int layer = 0; layer < 2; ++layer) {
    const float* W    = layer ? W1  : W0;
    const float* asv  = layer ? as1 : as0;
    const float* adv  = layer ? ad1 : ad0;
    const float* bias = layer ? b1  : b0;
    const float* gam  = layer ? g1  : g0;
    const float* bet  = layer ? be1 : be0;

    wt_bf16_kernel<<<cdiv(HID * HID, T), T, 0, stream>>>(W, WT);
    gemm_bf16_kernel<<<NN / 16, 256, 0, stream>>>(XB, WT, Hf);
    alpha_kernel<<<NN, 128, 0, stream>>>(Hf, asv, adv, AS, AD);

    zero_kernel<<<cdiv(NN * HH, T), T, 0, stream>>>((float*)MB, NN * HH);
    zero_kernel<<<cdiv(NN * HH, T), T, 0, stream>>>(DEN, NN * HH);
    zero_kernel<<<cdiv(NN * HID, T), T, 0, stream>>>(AGG, NN * HID);

    edge_logit_max_kernel<<<cdiv(ETOT * HH, T), T, 0, stream>>>(eidx, AS, AD, EV, MB);
    edge_exp_sum_kernel<<<cdiv(ETOT * HH, T), T, 0, stream>>>(eidx, MB, EV, DEN);
    edge_aggregate_kernel<<<cdiv(ETOT * HID, T), T, 0, stream>>>(eidx, Hf, EV, DEN, AGG);

    // bias + LN + ReLU; writes bf16 (next GEMM input) and f32 (pool input after layer 1)
    ln_relu_kernel<<<NN, 128, 0, stream>>>(AGG, bias, gam, bet, XB, Hf);
  }

  zero_kernel<<<cdiv(BB * HID, T), T, 0, stream>>>(SUM, BB * HID);
  zero_kernel<<<cdiv(BB, T), T, 0, stream>>>(CNT, BB);
  pool_scatter_kernel<<<cdiv(NN * HID, T), T, 0, stream>>>(Hf, batch, SUM, CNT);
  pool_final_kernel<<<cdiv(BB * HID, T), T, 0, stream>>>(SUM, CNT, (float*)d_out);
}
